// H3_60215441490629
// MI455X (gfx1250) — compile-verified, matrix operand order checked by reading, not run on hardware
//
#include <hip/hip_runtime.h>
#include <hip/hip_bf16.h>

// ---------------------------------------------------------------------------
// H3 layer on MI455X (gfx1250): bf16 WMMA for all matmul-like work.
//   B=4, L=2048, D=2048.
// GEMMs: double-buffered async global->LDS staging (ASYNCcnt), 128x128 block,
//        8 waves, each wave 32x64 via 2x4 WMMA fragments.
// Long conv: blocked-Toeplitz WMMA (causal), fused skip + gating.
// ---------------------------------------------------------------------------

#define B_SZ 4
#define L_SZ 2048
#define D_SZ 2048
#define PADV 256               // zero pad (elements) in front of each kv channel

typedef __attribute__((ext_vector_type(16))) __bf16 v16bf;
typedef __attribute__((ext_vector_type(8)))  __bf16 v8bf;
typedef __attribute__((ext_vector_type(8)))  float  v8f;

// ---- CDNA5 async global->LDS copy (bypasses VGPRs, tracked by ASYNCcnt) ----
__device__ __forceinline__ void async_ld16(unsigned ldsByteOff,
                                           const __bf16* gsrc) {
    // per-lane: LDS[ldsByteOff] <- 16 bytes at gsrc
    asm volatile("global_load_async_to_lds_b128 %0, %1, off"
                 :: "v"(ldsByteOff), "v"(gsrc)
                 : "memory");
}
__device__ __forceinline__ void wait_async0() {
    asm volatile("s_wait_asynccnt 0x0" ::: "memory");
}

// ------------------------------ fp32 -> bf16 --------------------------------
__global__ void f32_to_bf16_kernel(const float* __restrict__ src,
                                   __bf16* __restrict__ dst, int n) {
    int i = blockIdx.x * 256 + threadIdx.x;
    if (i < n) dst[i] = (__bf16)src[i];
}

// ---------------------------------------------------------------------------
// Build kerA: per (d, t, lane) one 16-element A-fragment slice for the
// Toeplitz WMMA.  r = 2t + (lane>>4), a = lane&15, element j is K=c:
//   kerA[((d*64+t)*32+lane)*16 + j] = ker[d][16r + a - j]  (0 if index < 0)
// ---------------------------------------------------------------------------
__global__ void build_kerA_kernel(const float* __restrict__ ker,
                                  __bf16* __restrict__ kerA) {
    int gi   = blockIdx.x * 256 + threadIdx.x;   // D*64*32 total
    int lane = gi & 31;
    int t    = (gi >> 5) & 63;
    int d    = gi >> 11;
    int r    = 2 * t + (lane >> 4);
    int a    = lane & 15;
    const float* kp = ker + (size_t)d * L_SZ;
    __bf16* op = kerA + (size_t)gi * 16;
#pragma unroll
    for (int j = 0; j < 16; ++j) {
        int kidx = 16 * r + a - j;               // always < L_SZ (r<=127,a<16)
        float v = (kidx >= 0) ? kp[kidx] : 0.0f;
        op[j] = (__bf16)v;
    }
}

// ---------------------------------------------------------------------------
// LDS-tiled bf16 WMMA GEMM:  C[m,n] = sum_k A[m,k]*W[n,k] + bias[n]
//   A: MxK row-major bf16, W: NxK row-major bf16 (i.e. u @ W^T).
// Block tile 128(M) x 128(N), K-step 32, double-buffered async staging.
// 8 waves arranged 4(M) x 2(N); each wave computes 32x64 = 2x4 fragments.
// TRANSPOSE_OUT: store bf16 to (B, N, Lb) layout (for q/k/v (B,D,L)),
// else store f32 row-major MxN (+bias) for the final projection.
// ---------------------------------------------------------------------------
template <bool TRANSPOSE_OUT>
__global__ __launch_bounds__(256)
void gemm_bf16_wmma(const __bf16* __restrict__ A,
                    const __bf16* __restrict__ W,
                    const float* __restrict__ bias,
                    void* __restrict__ Cout,
                    int M, int N, int K, int Lb) {
    __shared__ __bf16 As[2][128 * 32];   // 2 x 8 KB
    __shared__ __bf16 Ws[2][128 * 32];   // 2 x 8 KB

    int tid  = threadIdx.x;
    int lane = tid & 31;
    int wid  = tid >> 5;

    int bm = blockIdx.y * 128;
    int bn = blockIdx.x * 128;
    int wm = (wid >> 1) * 32;            // wave M offset: 0,32,64,96
    int wn = (wid & 1) * 64;             // wave N offset: 0,64

    // staging: thread -> (row = tid>>1, kcol = (tid&1)*16), 32B per matrix
    int srow = tid >> 1;
    int scol = (tid & 1) * 16;
    const __bf16* agp = A + (size_t)(bm + srow) * K + scol;
    const __bf16* wgp = W + (size_t)(bn + srow) * K + scol;
    unsigned aoff = (unsigned)(uintptr_t)(&As[0][srow * 32 + scol]);
    unsigned woff = (unsigned)(uintptr_t)(&Ws[0][srow * 32 + scol]);
    const unsigned bufStride = 128 * 32 * 2;   // bytes per buffer

    auto issue_tile = [&](int buf, int k0) {
        const __bf16* ga = agp + k0;
        const __bf16* gw = wgp + k0;
        unsigned la = aoff + (unsigned)buf * bufStride;
        unsigned lw = woff + (unsigned)buf * bufStride;
        async_ld16(la,      ga);
        async_ld16(la + 16, ga + 8);
        async_ld16(lw,      gw);
        async_ld16(lw + 16, gw + 8);
    };

    int fl = lane & 15;                  // fragment row/col for this lane
    int kb = (lane >> 4) * 16;           // K half selected by lane group

    v8f acc[2][4];
#pragma unroll
    for (int mi = 0; mi < 2; ++mi)
#pragma unroll
        for (int ni = 0; ni < 4; ++ni)
            acc[mi][ni] = (v8f){0.f, 0.f, 0.f, 0.f, 0.f, 0.f, 0.f, 0.f};

    const int nsteps = K / 32;
    issue_tile(0, 0);                    // prologue: tile 0 -> buffer 0
    for (int s = 0; s < nsteps; ++s) {
        int cur = s & 1;
        wait_async0();                   // my loads for buffer `cur` complete
        __syncthreads();                 // everyone's loads complete; buffer
                                         // cur^1 fully consumed by all waves
        if (s + 1 < nsteps) issue_tile(cur ^ 1, (s + 1) * 32);

        v16bf af[2], wf[4];
#pragma unroll
        for (int mi = 0; mi < 2; ++mi)
            af[mi] = *(const v16bf*)(&As[cur][(wm + mi * 16 + fl) * 32 + kb]);
#pragma unroll
        for (int ni = 0; ni < 4; ++ni)
            wf[ni] = *(const v16bf*)(&Ws[cur][(wn + ni * 16 + fl) * 32 + kb]);
#pragma unroll
        for (int mi = 0; mi < 2; ++mi)
#pragma unroll
            for (int ni = 0; ni < 4; ++ni)
                acc[mi][ni] = __builtin_amdgcn_wmma_f32_16x16x32_bf16(
                    false, af[mi], false, wf[ni], (short)0, acc[mi][ni],
                    false, false);
    }

    int hi = (lane >> 4) * 8;            // C-frag rows: r + hi
#pragma unroll
    for (int mi = 0; mi < 2; ++mi) {
#pragma unroll
        for (int ni = 0; ni < 4; ++ni) {
            int m0 = bm + wm + mi * 16;
            int n  = bn + wn + ni * 16 + fl;
            float bv = bias[n];
            if (TRANSPOSE_OUT) {
                // (B, N, Lb) bf16: 8 consecutive l per lane -> one 16B store
                __bf16* T = (__bf16*)Cout;
                int mrow0 = m0 + hi;
                int b  = mrow0 / Lb;
                int l0 = mrow0 % Lb;
                v8bf o;
#pragma unroll
                for (int r = 0; r < 8; ++r)
                    o[r] = (__bf16)(acc[mi][ni][r] + bv);
                *(v8bf*)(T + ((size_t)b * N + n) * Lb + l0) = o;
            } else {
                float* C = (float*)Cout;
#pragma unroll
                for (int r = 0; r < 8; ++r) {
                    int m = m0 + hi + r;
                    C[(size_t)m * N + n] = acc[mi][ni][r] + bv;
                }
            }
        }
    }
}

// ---------------------------------------------------------------------------
// Causal depthwise conv3 on q,k (+bias), kv = conv(k)*v, write:
//   qc  bf16 (B,D,L)
//   kvb bf16 (B,D,PADV+L) with PADV leading zeros per channel.
// grid = (9, B*D), block = 256   ->  pos in [0, 2304) = PADV + L exactly.
// ---------------------------------------------------------------------------
__global__ void shortconv_kv_kernel(const __bf16* __restrict__ qt,
                                    const __bf16* __restrict__ kt,
                                    const __bf16* __restrict__ vt,
                                    const float* __restrict__ cqw,
                                    const float* __restrict__ cqb,
                                    const float* __restrict__ ckw,
                                    const float* __restrict__ ckb,
                                    __bf16* __restrict__ qc,
                                    __bf16* __restrict__ kvb) {
    int ch  = blockIdx.y;                         // b*D + d
    int d   = ch & (D_SZ - 1);
    int pos = blockIdx.x * 256 + threadIdx.x;     // [0, PADV+L)
    size_t pbase = (size_t)ch * (L_SZ + PADV);
    if (pos < PADV) { kvb[pbase + pos] = (__bf16)0.0f; return; }
    int l = pos - PADV;
    size_t base = (size_t)ch * L_SZ;

    float q2 = (float)qt[base + l];
    float q1 = (l >= 1) ? (float)qt[base + l - 1] : 0.0f;
    float q0 = (l >= 2) ? (float)qt[base + l - 2] : 0.0f;
    float qv = cqw[d * 3 + 0] * q0 + cqw[d * 3 + 1] * q1 +
               cqw[d * 3 + 2] * q2 + cqb[d];

    float k2 = (float)kt[base + l];
    float k1 = (l >= 1) ? (float)kt[base + l - 1] : 0.0f;
    float k0 = (l >= 2) ? (float)kt[base + l - 2] : 0.0f;
    float kc = ckw[d * 3 + 0] * k0 + ckw[d * 3 + 1] * k1 +
               ckw[d * 3 + 2] * k2 + ckb[d];

    float kv = kc * (float)vt[base + l];
    qc[base + l]     = (__bf16)qv;
    kvb[pbase + pos] = (__bf16)kv;
}

// ---------------------------------------------------------------------------
// Causal long conv as blocked-Toeplitz WMMA, fused with D-skip + q gating.
// Wave handles (b, d, Phat):  Y[a,p] = sum_r K_r[a,c] * kv[16(p-r)+c],
// two r's packed per K=32 WMMA.  Epilogue: y = (y + kv*Dskip)*qc,
// scatter bf16 to yg (B, L, D).  grid = B*D blocks of 256 (8 waves).
// ---------------------------------------------------------------------------
__global__ __launch_bounds__(256)
void longconv_wmma_kernel(const __bf16* __restrict__ kvb,
                          const __bf16* __restrict__ kerA,
                          const __bf16* __restrict__ qc,
                          const float* __restrict__ Dskip,
                          __bf16* __restrict__ yg) {
    int lane = threadIdx.x & 31;
    int wid  = threadIdx.x >> 5;
    int wg   = blockIdx.x * 8 + wid;      // [0, B*D*8)
    int Ph   = wg & 7;
    int ch   = wg >> 3;                   // b*D + d
    int d    = ch & (D_SZ - 1);
    int b    = ch >> 11;

    const __bf16* kvp = kvb + (size_t)ch * (L_SZ + PADV) + PADV;
    const __bf16* kap = kerA + (size_t)d * (64 * 512) + lane * 16;

    int q   = 16 * Ph + (lane & 15);      // B-frag column (= output p)
    int rof = lane >> 4;                  // which r of the K=32 pair

    v8f acc = {0.f, 0.f, 0.f, 0.f, 0.f, 0.f, 0.f, 0.f};
    int tmax = 8 * Ph + 8;                // r in [0, 16*Ph+16) covers causality
    for (int t = 0; t < tmax; ++t) {
        v16bf af  = *(const v16bf*)(kap + (size_t)t * 512);
        int r = 2 * t + rof;
        v16bf bf_ = *(const v16bf*)(kvp + 16 * (q - r));  // pad makes r>q safe
        acc = __builtin_amdgcn_wmma_f32_16x16x32_bf16(
            false, af, false, bf_, (short)0, acc, false, false);
    }

    // epilogue: times i = 16*p + rof*8 + r  (8 contiguous per lane)
    int p  = q;
    int i0 = 16 * p + rof * 8;
    float ds = Dskip[d];
    v8bf kvv = *(const v8bf*)(kvp + i0);
    v8bf qv  = *(const v8bf*)(qc + (size_t)ch * L_SZ + i0);
    __bf16* op = yg + ((size_t)b * L_SZ + i0) * D_SZ + d;
#pragma unroll
    for (int r = 0; r < 8; ++r) {
        float y = acc[r] + (float)kvv[r] * ds;
        y *= (float)qv[r];
        op[(size_t)r * D_SZ] = (__bf16)y;
    }
}

// ---------------------------------------------------------------------------
extern "C" void kernel_launch(void* const* d_in, const int* in_sizes, int n_in,
                              void* d_out, int out_size, void* d_ws, size_t ws_size,
                              hipStream_t stream) {
    (void)in_sizes; (void)n_in; (void)out_size; (void)ws_size;
    const float* u     = (const float*)d_in[0];
    const float* Wq    = (const float*)d_in[1];
    const float* bq    = (const float*)d_in[2];
    const float* Wk    = (const float*)d_in[3];
    const float* bk    = (const float*)d_in[4];
    const float* Wv    = (const float*)d_in[5];
    const float* bv    = (const float*)d_in[6];
    const float* cqw   = (const float*)d_in[7];
    const float* cqb   = (const float*)d_in[8];
    const float* ckw   = (const float*)d_in[9];
    const float* ckb   = (const float*)d_in[10];
    const float* ssm   = (const float*)d_in[11];
    const float* Dskip = (const float*)d_in[12];
    const float* Wo    = (const float*)d_in[13];
    const float* bo    = (const float*)d_in[14];
    float* out = (float*)d_out;

    const int M = B_SZ * L_SZ;            // 8192
    const size_t nU = (size_t)M * D_SZ;   // 16.8M
    const size_t nW = (size_t)D_SZ * D_SZ;

    // bump allocator on d_ws (~407 MB total)
    char* ws = (char*)d_ws;
    size_t off = 0;
    auto alloc = [&](size_t bytes) -> void* {
        void* p = ws + off;
        off = (off + bytes + 255) & ~(size_t)255;
        return p;
    };
    __bf16* ub   = (__bf16*)alloc(nU * 2);
    __bf16* wqb  = (__bf16*)alloc(nW * 2);
    __bf16* wkb  = (__bf16*)alloc(nW * 2);
    __bf16* wvb  = (__bf16*)alloc(nW * 2);
    __bf16* wob  = (__bf16*)alloc(nW * 2);
    __bf16* qt   = (__bf16*)alloc(nU * 2);                     // (B,D,L)
    __bf16* kt   = (__bf16*)alloc(nU * 2);
    __bf16* vt   = (__bf16*)alloc(nU * 2);
    __bf16* qc   = (__bf16*)alloc(nU * 2);                     // (B,D,L)
    __bf16* kvb  = (__bf16*)alloc((size_t)B_SZ * D_SZ * (L_SZ + PADV) * 2);
    __bf16* kerA = (__bf16*)alloc((size_t)D_SZ * 64 * 32 * 16 * 2);  // 134 MB
    __bf16* yg   = (__bf16*)alloc(nU * 2);                     // (B,L,D)

    // 1) precision conversions + Toeplitz kernel-fragment build
    f32_to_bf16_kernel<<<(int)((nU + 255) / 256), 256, 0, stream>>>(u, ub, (int)nU);
    f32_to_bf16_kernel<<<(int)((nW + 255) / 256), 256, 0, stream>>>(Wq, wqb, (int)nW);
    f32_to_bf16_kernel<<<(int)((nW + 255) / 256), 256, 0, stream>>>(Wk, wkb, (int)nW);
    f32_to_bf16_kernel<<<(int)((nW + 255) / 256), 256, 0, stream>>>(Wv, wvb, (int)nW);
    f32_to_bf16_kernel<<<(int)((nW + 255) / 256), 256, 0, stream>>>(Wo, wob, (int)nW);
    build_kerA_kernel<<<(D_SZ * 64 * 32) / 256, 256, 0, stream>>>(ssm, kerA);

    // 2) q/k/v projections (WMMA), store transposed (B,D,L) bf16
    dim3 gemm_grid(D_SZ / 128, M / 128);   // (16, 64)
    gemm_bf16_wmma<true><<<gemm_grid, 256, 0, stream>>>(ub, wqb, bq, qt, M, D_SZ, D_SZ, L_SZ);
    gemm_bf16_wmma<true><<<gemm_grid, 256, 0, stream>>>(ub, wkb, bk, kt, M, D_SZ, D_SZ, L_SZ);
    gemm_bf16_wmma<true><<<gemm_grid, 256, 0, stream>>>(ub, wvb, bv, vt, M, D_SZ, D_SZ, L_SZ);

    // 3) depthwise causal conv3 + kv = conv(k)*v  (+ zero pad region)
    shortconv_kv_kernel<<<dim3((L_SZ + PADV) / 256, B_SZ * D_SZ), 256, 0, stream>>>(
        qt, kt, vt, cqw, cqb, ckw, ckb, qc, kvb);

    // 4) causal long conv (blocked Toeplitz WMMA) fused with skip + gating
    longconv_wmma_kernel<<<B_SZ * D_SZ, 256, 0, stream>>>(kvb, kerA, qc, Dskip, yg);

    // 5) output projection (WMMA) -> f32 (B,L,D)
    gemm_bf16_wmma<false><<<gemm_grid, 256, 0, stream>>>(yg, wob, bo, out, M, D_SZ, D_SZ, L_SZ);
}
